// PointNetSetAbstraction_22213570855529
// MI455X (gfx1250) — compile-verified, matrix operand order checked by reading, not run on hardware
//
#include <hip/hip_runtime.h>

// ---------------------------------------------------------------------------
// PointNet++ Set Abstraction for MI455X (gfx1250, wave32, WMMA)
//   FPS -> ball-query(+gather/concat, f16) -> 3x [WMMA GEMM + BN-stat] -> maxpool
// ---------------------------------------------------------------------------

typedef _Float16 h16;
typedef __attribute__((ext_vector_type(8)))  h16   h8;
typedef __attribute__((ext_vector_type(16))) h16   v16h;
typedef __attribute__((ext_vector_type(8)))  float v8f;

#define BATCH    8
#define NPTS     8192
#define CFEAT    64
#define NPOINT_  1024
#define NSAMPLE_ 32
#define RADIUS2  0.04f
#define BN_EPS_  1e-5f

constexpr int    RROWS = BATCH * NPOINT_ * NSAMPLE_;   // 262144 GEMM rows

// ---- workspace layout (bytes) ---------------------------------------------
constexpr size_t OFF_FPS   = 0;                                  // 8*1024*4
constexpr size_t OFF_WH0   = 32768;                              // 64*96*2
constexpr size_t OFF_WH1   = OFF_WH0 + 64 * 96 * 2;
constexpr size_t OFF_WH2   = OFF_WH1 + 64 * 64 * 2;
constexpr size_t OFF_STATS = OFF_WH2 + 128 * 64 * 2;             // 6*128 f32
constexpr size_t OFF_BN    = OFF_STATS + 6 * 128 * 4;            // 6*128 f32
constexpr size_t OFF_A     = ((OFF_BN + 6 * 128 * 4) + 255) & ~(size_t)255;
constexpr size_t SZ_A      = (size_t)RROWS * 96 * 2;             // 48 MB (X0 / Y1)
constexpr size_t OFF_B     = OFF_A + SZ_A;                       // 64 MB (Y0 / Y2)
// total ws need: ~118 MB

__device__ inline unsigned ballot32(bool p) {
#if __has_builtin(__builtin_amdgcn_ballot_w32)
    return __builtin_amdgcn_ballot_w32(p);
#else
    return (unsigned)__ballot((int)p);
#endif
}

// ---------------------------------------------------------------------------
// 1) Farthest point sampling: one block per batch, point cache in VGPRs,
//    argmax via LDS tree reduction (tie-break = lowest index, like argmax).
// ---------------------------------------------------------------------------
__global__ void fps_kernel(const float* __restrict__ xyz,
                           float* __restrict__ out_xyz,
                           int* __restrict__ fps_idx) {
    __shared__ float rv[256];
    __shared__ int   ri[256];
    const int b   = blockIdx.x;
    const int tid = threadIdx.x;
    const float* X = xyz + (size_t)b * NPTS * 3;

    float d[32], px[32], py[32], pz[32];
#pragma unroll
    for (int k = 0; k < 32; ++k) {
        int j = k * 256 + tid;
        px[k] = X[j * 3 + 0];
        py[k] = X[j * 3 + 1];
        pz[k] = X[j * 3 + 2];
        d[k]  = 1e10f;
    }

    int far = 0;
    for (int i = 0; i < NPOINT_; ++i) {
        float cx = X[far * 3 + 0], cy = X[far * 3 + 1], cz = X[far * 3 + 2];
        if (tid == 0) {
            fps_idx[b * NPOINT_ + i] = far;
            float* o = out_xyz + ((size_t)b * NPOINT_ + i) * 3;
            o[0] = cx; o[1] = cy; o[2] = cz;
        }
        float bestv = -1.0f;
        int   bestj = 0;
#pragma unroll
        for (int k = 0; k < 32; ++k) {
            float dx = px[k] - cx, dy = py[k] - cy, dz = pz[k] - cz;
            float dd = dx * dx + dy * dy + dz * dz;
            d[k] = fminf(d[k], dd);
            if (d[k] > bestv) { bestv = d[k]; bestj = k * 256 + tid; }
        }
        rv[tid] = bestv;
        ri[tid] = bestj;
        __syncthreads();
        for (int s = 128; s > 0; s >>= 1) {
            if (tid < s) {
                float v2 = rv[tid + s];
                int   i2 = ri[tid + s];
                if (v2 > rv[tid] || (v2 == rv[tid] && i2 < ri[tid])) {
                    rv[tid] = v2; ri[tid] = i2;
                }
            }
            __syncthreads();
        }
        far = ri[0];
        __syncthreads();
    }
}

// ---------------------------------------------------------------------------
// 2) Ball query (wave32 ballot + prefix-popc ordered selection) fused with
//    gather + concat. Writes X0 row-major (R x 96) f16, cols: 3 xyz-diff,
//    64 features, 29 zero pad (so K is a multiple of 32 for WMMA).
// ---------------------------------------------------------------------------
__global__ void ballquery_gather_kernel(const float* __restrict__ xyz,
                                        const float* __restrict__ feat,
                                        const float* __restrict__ new_xyz,
                                        h16* __restrict__ X0) {
    __shared__ int sidx[8][NSAMPLE_];
    const int w    = threadIdx.x >> 5;
    const int lane = threadIdx.x & 31;
    const int q    = blockIdx.x * 8 + w;            // 0..8191 query points
    const int b    = q >> 10;
    const float* X = xyz + (size_t)b * NPTS * 3;
    const float cx = new_xyz[(size_t)q * 3 + 0];
    const float cy = new_xyz[(size_t)q * 3 + 1];
    const float cz = new_xyz[(size_t)q * 3 + 2];

    int cnt = 0;
    for (int base = 0; base < NPTS; base += 32) {
        if (cnt >= NSAMPLE_) break;                 // uniform across wave
        int   j  = base + lane;
        float dx = X[j * 3 + 0] - cx;
        float dy = X[j * 3 + 1] - cy;
        float dz = X[j * 3 + 2] - cz;
        bool  inb = (dx * dx + dy * dy + dz * dz) <= RADIUS2;
        unsigned m   = ballot32(inb);
        int      pos = cnt + __popc(m & ((1u << lane) - 1u));
        if (inb && pos < NSAMPLE_) sidx[w][pos] = j;
        cnt += __popc(m);
    }
    __syncthreads();
    if (lane >= cnt && lane < NSAMPLE_) sidx[w][lane] = sidx[w][0]; // pad w/ first
    __syncthreads();

    // gather: lane k handles sample k of its wave's query point
    int    j   = sidx[w][lane];
    size_t r   = (size_t)q * NSAMPLE_ + lane;
    h16*   row = X0 + r * 96;
    row[0] = (h16)(X[j * 3 + 0] - cx);
    row[1] = (h16)(X[j * 3 + 1] - cy);
    row[2] = (h16)(X[j * 3 + 2] - cz);
    const float* F = feat + (size_t)b * CFEAT * NPTS + j;
#pragma unroll
    for (int c = 0; c < CFEAT; ++c) row[3 + c] = (h16)F[(size_t)c * NPTS];
#pragma unroll
    for (int c = 67; c < 96; ++c) row[c] = (h16)0.0f;
}

// ---------------------------------------------------------------------------
// 3) Weight f32->f16 (+pad K 67->96) and zero the BN stat accumulators.
// ---------------------------------------------------------------------------
__global__ void prep_kernel(const float* __restrict__ W0,
                            const float* __restrict__ W1,
                            const float* __restrict__ W2,
                            h16* __restrict__ Wh0, h16* __restrict__ Wh1,
                            h16* __restrict__ Wh2, float* __restrict__ stats) {
    int t = threadIdx.x;
    for (int i = t; i < 64 * 96; i += 256) {
        int o = i / 96, c = i % 96;
        Wh0[i] = (c < 67) ? (h16)W0[o * 67 + c] : (h16)0.0f;
    }
    for (int i = t; i < 64 * 64; i += 256) Wh1[i] = (h16)W1[i];
    for (int i = t; i < 128 * 64; i += 256) Wh2[i] = (h16)W2[i];
    for (int i = t; i < 6 * 128; i += 256) stats[i] = 0.0f;
}

// ---------------------------------------------------------------------------
// 4) WMMA GEMM: Y = act(in) * W^T + bias, plus per-channel sum/sumsq atomics.
//    MODE 0: in row-major (R x CIN) f16, no BN     (layer 0)
//    MODE 1: in col-major (CIN x R) f16, BN+ReLU on load (layers 1,2)
//    Output: col-major (COUT x R) f16 -> contiguous b128 D-fragment stores.
//    One wave per 16x16 tile; wave id == output-column tile; 64 row tiles per
//    block so the stat atomics are amortized (~65k total).
// ---------------------------------------------------------------------------
template <int MODE, int CIN, int COUT>
__global__ void gemm_bn_kernel(const h16* __restrict__ in,
                               const float* __restrict__ bnscale,
                               const float* __restrict__ bnshift,
                               const h16* __restrict__ Wh,
                               const float* __restrict__ bias,
                               h16* __restrict__ out,
                               float* __restrict__ ssum,
                               float* __restrict__ ssq) {
    const int lane = threadIdx.x & 31;
    const int wv   = threadIdx.x >> 5;        // wave id == column tile
    const int n0   = wv * 16;
    const int m    = lane & 15;               // A row / D column lane index
    const int kb   = (lane >> 4) * 8;         // K-half select (ISA A layout)
    const int ncol = n0 + m;                  // this lane's output channel
    const float bv = bias[ncol];
    float s1 = 0.0f, s2 = 0.0f;

    // B fragments (W^T): identical lane layout to A with row -> out-channel.
    v16h bfrag[CIN / 32];
#pragma unroll
    for (int kt = 0; kt < CIN / 32; ++kt) {
        const h16* p = Wh + (size_t)ncol * CIN + kt * 32 + kb;
        h8 lo = *(const h8*)p;
        h8 hi = *(const h8*)(p + 16);
        v16h tb;
#pragma unroll
        for (int i = 0; i < 8; ++i) { tb[i] = lo[i]; tb[8 + i] = hi[i]; }
        bfrag[kt] = tb;
    }

    // Hoist BN scale/shift for this lane's K slots (constant across row tiles)
    float scl[CIN / 32][16], shf[CIN / 32][16];
    if (MODE == 1) {
#pragma unroll
        for (int kt = 0; kt < CIN / 32; ++kt)
#pragma unroll
            for (int i = 0; i < 8; ++i) {
                int c = kt * 32 + kb + i;
                scl[kt][i] = bnscale[c];      shf[kt][i] = bnshift[c];
                scl[kt][8 + i] = bnscale[c + 16]; shf[kt][8 + i] = bnshift[c + 16];
            }
    }

    for (int tIt = 0; tIt < 64; ++tIt) {
        const int r0 = (blockIdx.x * 64 + tIt) * 16;
        v8f acc = {0.f, 0.f, 0.f, 0.f, 0.f, 0.f, 0.f, 0.f};
#pragma unroll
        for (int kt = 0; kt < CIN / 32; ++kt) {
            v16h a;
            if (MODE == 0) {
                const h16* p = in + (size_t)(r0 + m) * CIN + kt * 32 + kb;
                __builtin_prefetch(p + 16 * CIN, 0, 1);   // next row tile
                h8 lo = *(const h8*)p;
                h8 hi = *(const h8*)(p + 16);
#pragma unroll
                for (int i = 0; i < 8; ++i) { a[i] = lo[i]; a[8 + i] = hi[i]; }
            } else {
#pragma unroll
                for (int i = 0; i < 8; ++i) {
                    int   c  = kt * 32 + kb + i;
                    float y  = (float)in[(size_t)c * RROWS + r0 + m]
                               * scl[kt][i] + shf[kt][i];
                    float y2 = (float)in[(size_t)(c + 16) * RROWS + r0 + m]
                               * scl[kt][8 + i] + shf[kt][8 + i];
                    a[i]     = (h16)fmaxf(y, 0.0f);
                    a[8 + i] = (h16)fmaxf(y2, 0.0f);
                }
            }
            acc = __builtin_amdgcn_wmma_f32_16x16x32_f16(
                false, a, false, bfrag[kt], (short)0, acc, false, false);
        }
        // epilogue: bias, BN statistics, f16 col-major store (b128 per lane)
        h8 hv;
#pragma unroll
        for (int j = 0; j < 8; ++j) {
            float v = acc[j] + bv;
            s1 += v;
            s2 += v * v;
            hv[j] = (h16)v;
        }
        const int rbase = r0 + ((lane >> 4) << 3);
        *(h8*)(out + (size_t)ncol * RROWS + rbase) = hv;
    }

    // fold the two lane-halves (same channel) and flush stats
    s1 += __shfl_xor(s1, 16, 32);
    s2 += __shfl_xor(s2, 16, 32);
    if (lane < 16) {
        atomicAdd(&ssum[ncol], s1);
        atomicAdd(&ssq[ncol], s2);
    }
}

// ---------------------------------------------------------------------------
// 5) Convert per-channel (sum, sumsq) into affine BN scale/shift.
// ---------------------------------------------------------------------------
__global__ void finalize_bn_kernel(const float* __restrict__ ssum,
                                   const float* __restrict__ ssq,
                                   const float* __restrict__ g,
                                   const float* __restrict__ beta,
                                   float* __restrict__ scale,
                                   float* __restrict__ shift, int cout) {
    int o = threadIdx.x;
    if (o < cout) {
        const float invn = 1.0f / (float)RROWS;
        float mean = ssum[o] * invn;
        float var  = ssq[o] * invn - mean * mean;
        float sc   = g[o] * rsqrtf(var + BN_EPS_);
        scale[o] = sc;
        shift[o] = beta[o] - mean * sc;
    }
}

// ---------------------------------------------------------------------------
// 6) Final BN + ReLU + max over the 32 samples -> new_features (B,128,1024).
//    Y2 is col-major so the 32 samples are contiguous f16 (b128 loads).
// ---------------------------------------------------------------------------
__global__ void maxpool_kernel(const h16* __restrict__ Y2,
                               const float* __restrict__ scale,
                               const float* __restrict__ shift,
                               float* __restrict__ out) {
    int t = blockIdx.x * 256 + threadIdx.x;      // (b*128+o)*1024 + s
    int s = t & 1023;
    int o = (t >> 10) & 127;
    int b = t >> 17;
    const h16* p  = Y2 + (size_t)o * RROWS + ((size_t)(b * 1024 + s)) * 32;
    const float sc = scale[o], sh = shift[o];
    float mx = 0.0f;                             // relu floor
#pragma unroll
    for (int k = 0; k < 32; ++k) {
        float v = (float)p[k] * sc + sh;
        mx = fmaxf(mx, v);
    }
    out[t] = mx;
}

// ---------------------------------------------------------------------------
extern "C" void kernel_launch(void* const* d_in, const int* in_sizes, int n_in,
                              void* d_out, int out_size, void* d_ws,
                              size_t ws_size, hipStream_t stream) {
    (void)in_sizes; (void)n_in; (void)out_size; (void)ws_size;
    const float* xyz  = (const float*)d_in[0];
    const float* feat = (const float*)d_in[1];
    const float* W0 = (const float*)d_in[2];
    const float* b0 = (const float*)d_in[3];
    const float* g0 = (const float*)d_in[4];
    const float* e0 = (const float*)d_in[5];
    const float* W1 = (const float*)d_in[6];
    const float* b1 = (const float*)d_in[7];
    const float* g1 = (const float*)d_in[8];
    const float* e1 = (const float*)d_in[9];
    const float* W2 = (const float*)d_in[10];
    const float* b2 = (const float*)d_in[11];
    const float* g2 = (const float*)d_in[12];
    const float* e2 = (const float*)d_in[13];

    float* out     = (float*)d_out;              // [new_xyz | new_features]
    char*  ws      = (char*)d_ws;
    int*   fpsIdx  = (int*)(ws + OFF_FPS);
    h16*   Wh0     = (h16*)(ws + OFF_WH0);
    h16*   Wh1     = (h16*)(ws + OFF_WH1);
    h16*   Wh2     = (h16*)(ws + OFF_WH2);
    float* stats   = (float*)(ws + OFF_STATS);   // 3 layers x (sum[128], sq[128])
    float* bn      = (float*)(ws + OFF_BN);      // 3 layers x (scale[128], shift[128])
    h16*   bufA    = (h16*)(ws + OFF_A);         // X0 (Rx96) then Y1 (64xR)
    h16*   bufB    = (h16*)(ws + OFF_B);         // Y0 (64xR) then Y2 (128xR)

    fps_kernel<<<BATCH, 256, 0, stream>>>(xyz, out, fpsIdx);
    ballquery_gather_kernel<<<1024, 256, 0, stream>>>(xyz, feat, out, bufA);
    prep_kernel<<<1, 256, 0, stream>>>(W0, W1, W2, Wh0, Wh1, Wh2, stats);

    // layer 0: (R x 96) x (96 x 64)
    gemm_bn_kernel<0, 96, 64><<<256, 128, 0, stream>>>(
        bufA, nullptr, nullptr, Wh0, b0, bufB, stats + 0, stats + 128);
    finalize_bn_kernel<<<1, 128, 0, stream>>>(stats + 0, stats + 128,
                                              g0, e0, bn + 0, bn + 128, 64);
    // layer 1: (R x 64) x (64 x 64), BN0+ReLU on load
    gemm_bn_kernel<1, 64, 64><<<256, 128, 0, stream>>>(
        bufB, bn + 0, bn + 128, Wh1, b1, bufA, stats + 256, stats + 384);
    finalize_bn_kernel<<<1, 128, 0, stream>>>(stats + 256, stats + 384,
                                              g1, e1, bn + 256, bn + 384, 64);
    // layer 2: (R x 64) x (64 x 128), BN1+ReLU on load
    gemm_bn_kernel<1, 64, 128><<<256, 256, 0, stream>>>(
        bufA, bn + 256, bn + 384, Wh2, b2, bufB, stats + 512, stats + 640);
    finalize_bn_kernel<<<1, 128, 0, stream>>>(stats + 512, stats + 640,
                                              g2, e2, bn + 512, bn + 640, 128);
    // BN2 + ReLU + max over samples
    maxpool_kernel<<<4096, 256, 0, stream>>>(bufB, bn + 512, bn + 640,
                                             out + BATCH * NPOINT_ * 3);
}
